// NMT_26336739459406
// MI455X (gfx1250) — matrix operand order repaired and checked by reading, not compiled, not run on hardware
//
#include <hip/hip_runtime.h>
#include <hip/hip_bf16.h>
#include <math.h>

#define DEV __device__ __forceinline__

typedef __attribute__((ext_vector_type(8)))  float          v8f;
typedef __attribute__((ext_vector_type(16))) __bf16         v16bf;
typedef __attribute__((ext_vector_type(16))) unsigned short u16x16;
typedef __attribute__((ext_vector_type(8)))  unsigned short u16x8;

constexpr int Hh = 1024, Ee = 512, Vv = 32000, Ss = 64, Tt = 64, Bb = 32;
constexpr int G4 = 4 * Hh;     // 4096 (gate width)
constexpr int FG = 500;        // final-gemm column groups (64 cols each): 32000/64

// ---------- scalar helpers ----------
DEV unsigned short f2bf(float f) {              // fp32 -> bf16 RNE
  unsigned int u = __builtin_bit_cast(unsigned int, f);
  u += 0x7fffu + ((u >> 16) & 1u);
  return (unsigned short)(u >> 16);
}
DEV float bf2f(unsigned short v) {
  unsigned int u = ((unsigned int)v) << 16;
  return __builtin_bit_cast(float, u);
}
DEV float sigm(float x) { return 1.0f / (1.0f + expf(-x)); }

// One lane's A-fragment (16x32 bf16, contiguous-k layout): 16 contiguous
// bf16 at arow[kbase..+15] -> two b128 loads.
DEV v16bf load_a_bf(const unsigned short* arow, int kbase) {
  u16x8 lo = *(const u16x8*)(arow + kbase);
  u16x8 hi = *(const u16x8*)(arow + kbase + 8);
  u16x16 u;
  #pragma unroll
  for (int i = 0; i < 8; ++i) { u[i] = lo[i]; u[8 + i] = hi[i]; }
  return __builtin_bit_cast(v16bf, u);
}

// Accumulate one "A[M,K] x W[N,K]^T" part into the wave's 16x16 f32 tile.
// A and W both bf16 row-major. m/n/half precomputed per lane.
DEV void gemm_part(v8f& acc, const unsigned short* A, int lda,
                   const unsigned short* W, int ldw, int K,
                   int m, int n, int half) {
  if (A == nullptr || K <= 0) return;
  const unsigned short* arow = A + (size_t)m * lda + half * 16;
  const unsigned short* wrow = W + (size_t)n * ldw + half * 16;
  for (int k0 = 0; k0 < K; k0 += 32) {
    v16bf av = load_a_bf(arow, k0);
    v16bf bv = *(const v16bf*)(wrow + k0);   // 16 contiguous bf16 of weight row n
    acc = __builtin_amdgcn_wmma_f32_16x16x32_bf16(
        false, av, false, bv, (short)0, acc, false, false);
  }
}

// ---------- generic WMMA GEMM: C = act( sum_parts A_i * W_i^T + bias ) ----------
// One wave per 16x16 output tile; grid supplies exactly tiles_m*tiles_n waves.
// Dual nullable outputs: fp32 Cf and/or bf16 Cbf.
__global__ void __launch_bounds__(256)
gemm_bf16_wmma(const unsigned short* A1, int lda1, const unsigned short* W1, int ldw1, int K1,
               const unsigned short* A2, int lda2, const unsigned short* W2, int ldw2, int K2,
               const unsigned short* A3, int lda3, const unsigned short* W3, int ldw3, int K3,
               const float* bias, float* Cf, unsigned short* Cbf,
               int ldc, int tiles_m, int act) {
  int wid  = (blockIdx.x * blockDim.x + threadIdx.x) >> 5;
  int lane = threadIdx.x & 31;
  int tm = wid % tiles_m, tn = wid / tiles_m;
  int m0 = tm * 16, n0 = tn * 16;
  int nl = lane & 15, half = lane >> 4;
  int m = m0 + nl, n = n0 + nl;

  v8f acc;
  #pragma unroll
  for (int r = 0; r < 8; ++r) acc[r] = 0.0f;

  gemm_part(acc, A1, lda1, W1, ldw1, K1, m, n, half);
  gemm_part(acc, A2, lda2, W2, ldw2, K2, m, n, half);
  gemm_part(acc, A3, lda3, W3, ldw3, K3, m, n, half);

  float bn = bias ? bias[n] : 0.0f;
  #pragma unroll
  for (int r = 0; r < 8; ++r) {
    float v = acc[r] + bn;
    if (act) v = tanhf(v);
    int row = m0 + r + half * 8;        // C/D layout: lanes 16-31 hold rows +8
    size_t idx = (size_t)row * ldc + n;
    if (Cf)  Cf[idx]  = v;
    if (Cbf) Cbf[idx] = f2bf(v);
  }
}

// ---------- LSTM cell (torch gate order i,f,g,o), optional length-mask ----------
// Writes fp32 state h/c plus a bf16 mirror of h (GEMM A operand).
__global__ void __launch_bounds__(256)
lstm_cell(const float* __restrict__ z, float* h, float* c, unsigned short* hbf,
          const int* enc_len, int t, float* out_store) {
  int idx = blockIdx.x * blockDim.x + threadIdx.x;   // over Bb*Hh
  int b = idx >> 10, hh = idx & (Hh - 1);
  const float* zb = z + (size_t)b * G4;
  float ig = sigm(zb[hh]);
  float fg = sigm(zb[Hh + hh]);
  float gg = tanhf(zb[2 * Hh + hh]);
  float og = sigm(zb[3 * Hh + hh]);
  float cp = c[idx];
  float c2 = fg * cp + ig * gg;
  float h2 = og * tanhf(c2);
  bool valid = enc_len ? (t < enc_len[b]) : true;
  float hold = h[idx];
  float hf = valid ? h2 : hold;
  h[idx] = hf;
  hbf[idx] = f2bf(hf);
  c[idx] = valid ? c2 : cp;
  if (out_store) out_store[idx] = valid ? h2 : 0.0f;  // pad_packed zeros at pads
}

// ---------- pt = sigmoid( tanh(yt W^T) . w ) * len   (tmp already tanh'd) ----------
__global__ void __launch_bounds__(256)
pt_kernel(const float* __restrict__ tmp, const float* __restrict__ w,
          const int* __restrict__ elen, float* pt) {
  int b = blockIdx.x, tid = threadIdx.x;
  __shared__ float sm[256];
  const float* r = tmp + (size_t)b * Hh;
  float s = 0.0f;
  for (int i = tid; i < Hh; i += 256) s += r[i] * w[i];
  sm[tid] = s; __syncthreads();
  for (int k = 128; k > 0; k >>= 1) { if (tid < k) sm[tid] += sm[tid + k]; __syncthreads(); }
  if (tid == 0) pt[b] = sigm(sm[0]) * (float)elen[b];
}

// ---------- local attention with Gaussian window; ct = (softmax*window) . enc_h ----------
// ct emitted directly in bf16 (consumed only as a GEMM A operand).
__global__ void __launch_bounds__(256)
attention(const float* __restrict__ yt, const float* __restrict__ enc_h,
          const float* __restrict__ pt, const int* __restrict__ elen,
          unsigned short* __restrict__ ct) {
  int b = blockIdx.x;
  int tid = threadIdx.x, lane = tid & 31, w = tid >> 5;  // 8 waves
  __shared__ float sc[Ss];
  const float* y = yt + (size_t)b * Hh;
  for (int s = w; s < Ss; s += 8) {
    const float* e = enc_h + ((size_t)s * Bb + b) * Hh;
    float d = 0.0f;
    for (int i = lane; i < Hh; i += 32) d += y[i] * e[i];
    for (int off = 16; off > 0; off >>= 1) d += __shfl_down(d, off, 32);
    if (lane == 0) sc[s] = d;
  }
  __syncthreads();
  if (tid == 0) {            // deterministic serial softmax over 64 entries
    int L = elen[b];
    float mx = -INFINITY;
    for (int s = 0; s < L; ++s) mx = fmaxf(mx, sc[s]);
    float sum = 0.0f;
    for (int s = 0; s < L; ++s) { float e = expf(sc[s] - mx); sc[s] = e; sum += e; }
    float p = pt[b];
    for (int s = 0; s < Ss; ++s) {
      float al = (s < L) ? sc[s] / sum : 0.0f;
      float d = (float)s - p;
      sc[s] = al * expf(-(d * d) * (1.0f / 50.0f));   // WIN_D^2/2 = 50
    }
  }
  __syncthreads();
  for (int h = tid; h < Hh; h += 256) {
    float a = 0.0f;
    for (int s = 0; s < Ss; ++s) a += sc[s] * enc_h[((size_t)s * Bb + b) * Hh + h];
    ct[(size_t)b * Hh + h] = f2bf(a);
  }
}

// ---------- final projection: per-wave 16x64 tile, emit per-row max/sumexp partials ----------
__global__ void __launch_bounds__(256)
final_gemm_partial(const unsigned short* __restrict__ A,
                   const unsigned short* __restrict__ Wf,
                   float* __restrict__ pmax, float* __restrict__ psum) {
  int wid  = (blockIdx.x * blockDim.x + threadIdx.x) >> 5;
  int lane = threadIdx.x & 31;
  int tm = wid % (Tt * Bb / 16);      // 128 row tiles
  int g  = wid / (Tt * Bb / 16);      // 500 column groups of 64
  int m0 = tm * 16, n0 = g * 64;
  int nl = lane & 15, half = lane >> 4;
  const unsigned short* arow = A + (size_t)(m0 + nl) * Hh + half * 16;

  v8f acc[4];
  #pragma unroll
  for (int j = 0; j < 4; ++j)
    #pragma unroll
    for (int r = 0; r < 8; ++r) acc[j][r] = 0.0f;

  const unsigned short* wr[4];
  #pragma unroll
  for (int j = 0; j < 4; ++j)
    wr[j] = Wf + (size_t)(n0 + j * 16 + nl) * Hh + half * 16;

  for (int k0 = 0; k0 < Hh; k0 += 32) {
    v16bf av = load_a_bf(arow, k0);
    #pragma unroll
    for (int j = 0; j < 4; ++j) {
      v16bf bv = *(const v16bf*)(wr[j] + k0);
      acc[j] = __builtin_amdgcn_wmma_f32_16x16x32_bf16(
          false, av, false, bv, (short)0, acc[j], false, false);
    }
  }
  // per-row (16 lanes of one half) max + sumexp over this wave's 64 columns
  #pragma unroll
  for (int r = 0; r < 8; ++r) {
    float lm = fmaxf(fmaxf(acc[0][r], acc[1][r]), fmaxf(acc[2][r], acc[3][r]));
    for (int off = 1; off < 16; off <<= 1) lm = fmaxf(lm, __shfl_xor(lm, off, 16));
    float ls = expf(acc[0][r] - lm) + expf(acc[1][r] - lm) +
               expf(acc[2][r] - lm) + expf(acc[3][r] - lm);
    for (int off = 1; off < 16; off <<= 1) ls += __shfl_xor(ls, off, 16);
    if (nl == 0) {
      int row = m0 + r + half * 8;
      pmax[(size_t)row * FG + g] = lm;
      psum[(size_t)row * FG + g] = ls;
    }
  }
}

__global__ void __launch_bounds__(256)
logz_combine(const float* __restrict__ pmax, const float* __restrict__ psum,
             float* __restrict__ logZ) {
  int row = blockIdx.x, tid = threadIdx.x;
  __shared__ float sm[256];
  float lm = -INFINITY;
  for (int g = tid; g < FG; g += 256) lm = fmaxf(lm, pmax[(size_t)row * FG + g]);
  sm[tid] = lm; __syncthreads();
  for (int s = 128; s > 0; s >>= 1) { if (tid < s) sm[tid] = fmaxf(sm[tid], sm[tid + s]); __syncthreads(); }
  float M = sm[0]; __syncthreads();
  float ls = 0.0f;
  for (int g = tid; g < FG; g += 256)
    ls += psum[(size_t)row * FG + g] * expf(pmax[(size_t)row * FG + g] - M);
  sm[tid] = ls; __syncthreads();
  for (int s = 128; s > 0; s >>= 1) { if (tid < s) sm[tid] += sm[tid + s]; __syncthreads(); }
  if (tid == 0) logZ[row] = M + logf(sm[0]);
}

__global__ void __launch_bounds__(256)
target_logit(const unsigned short* __restrict__ dec_out,
             const unsigned short* __restrict__ Wf,
             const int* __restrict__ tgt, float* __restrict__ tl) {
  int blk = blockIdx.x;                 // (Tt-1)*Bb blocks
  int t = blk >> 5, b = blk & 31;
  int row = t * Bb + b;
  int tok = tgt[(t + 1) * Bb + b];
  const unsigned short* a = dec_out + (size_t)row * Hh;
  const unsigned short* w = Wf + (size_t)tok * Hh;
  int tid = threadIdx.x;
  __shared__ float sm[256];
  float s = 0.0f;
  for (int i = tid; i < Hh; i += 256) s += bf2f(a[i]) * bf2f(w[i]);
  sm[tid] = s; __syncthreads();
  for (int k = 128; k > 0; k >>= 1) { if (tid < k) sm[tid] += sm[tid + k]; __syncthreads(); }
  if (tid == 0) tl[row] = sm[0];
}

__global__ void __launch_bounds__(64)
sum_out(const float* __restrict__ tl, const float* __restrict__ logZ,
        const int* __restrict__ tgt, float* __restrict__ out) {
  int b = blockIdx.x, tt = threadIdx.x;      // 64 threads, tt==63 contributes 0
  __shared__ float sm[64];
  float v = 0.0f;
  if (tt < Tt - 1) {
    int row = tt * Bb + b;
    float mask = (tgt[(tt + 1) * Bb + b] != 0) ? 1.0f : 0.0f;
    v = (tl[row] - logZ[row]) * mask;
  }
  sm[tt] = v; __syncthreads();
  for (int s = 32; s > 0; s >>= 1) { if (tt < s) sm[tt] += sm[tt + s]; __syncthreads(); }
  if (tt == 0) out[b] = sm[0];
}

// ---------- small utility kernels ----------
__global__ void to_bf16(const float* __restrict__ s, unsigned short* __restrict__ d, int n) {
  int i = blockIdx.x * blockDim.x + threadIdx.x;
  if (i < n) d[i] = f2bf(s[i]);
}
__global__ void gather_emb_bf(const int* __restrict__ tok, const float* __restrict__ emb,
                              unsigned short* __restrict__ out, int n) {
  int i = blockIdx.x * blockDim.x + threadIdx.x;
  if (i < n) out[i] = f2bf(emb[(size_t)tok[i >> 9] * Ee + (i & 511)]);
}
__global__ void zero_f32(float* p, int n) {
  int i = blockIdx.x * blockDim.x + threadIdx.x;
  if (i < n) p[i] = 0.0f;
}

// =======================================================================
extern "C" void kernel_launch(void* const* d_in, const int* in_sizes, int n_in,
                              void* d_out, int out_size, void* d_ws, size_t ws_size,
                              hipStream_t stream) {
  (void)in_sizes; (void)n_in; (void)out_size; (void)ws_size;
  const int*   src_tok = (const int*)d_in[0];
  const int*   tgt_tok = (const int*)d_in[1];
  const int*   elen    = (const int*)d_in[2];
  const float* emb_src = (const float*)d_in[3];
  const float* emb_tar = (const float*)d_in[4];
  const float* eW_ih0 = (const float*)d_in[5];
  const float* eW_hh0 = (const float*)d_in[6];
  const float* eb0    = (const float*)d_in[7];
  const float* eW_ih1 = (const float*)d_in[8];
  const float* eW_hh1 = (const float*)d_in[9];
  const float* eb1    = (const float*)d_in[10];
  const float* dW_ih0 = (const float*)d_in[11];
  const float* dW_hh0 = (const float*)d_in[12];
  const float* db0    = (const float*)d_in[13];
  const float* dW_ih1 = (const float*)d_in[14];
  const float* dW_hh1 = (const float*)d_in[15];
  const float* db1    = (const float*)d_in[16];
  const float* W_h2t  = (const float*)d_in[17];
  const float* w_t2p  = (const float*)d_in[18];
  const float* W_c2h  = (const float*)d_in[19];
  const float* W_fin  = (const float*)d_in[20];

  // --- workspace bump allocator (256B aligned) ---
  char* base = (char*)d_ws; size_t off = 0;
  auto alloc = [&](size_t bytes) -> void* {
    off = (off + 255) & ~(size_t)255;
    void* r = base + off; off += bytes; return r;
  };
  auto allocBF = [&](size_t elems) { return (unsigned short*)alloc(elems * 2); };
  auto allocF  = [&](size_t elems) { return (float*)alloc(elems * 4); };

  unsigned short* bWih0e = allocBF((size_t)G4 * Ee);
  unsigned short* bWhh0e = allocBF((size_t)G4 * Hh);
  unsigned short* bWih1e = allocBF((size_t)G4 * Hh);
  unsigned short* bWhh1e = allocBF((size_t)G4 * Hh);
  unsigned short* bWih0d = allocBF((size_t)G4 * (Ee + Hh));
  unsigned short* bWhh0d = allocBF((size_t)G4 * Hh);
  unsigned short* bWih1d = allocBF((size_t)G4 * Hh);
  unsigned short* bWhh1d = allocBF((size_t)G4 * Hh);
  unsigned short* bWh2t  = allocBF((size_t)Hh * Hh);
  unsigned short* bWc2h  = allocBF((size_t)Hh * 2 * Hh);
  unsigned short* bWfin  = allocBF((size_t)Vv * Hh);

  unsigned short* x_src_bf   = allocBF((size_t)Ss * Bb * Ee);
  unsigned short* y_tar_bf   = allocBF((size_t)Tt * Bb * Ee);
  unsigned short* dec_out_bf = allocBF((size_t)Tt * Bb * Hh);
  unsigned short* bfmir      = allocBF((size_t)3 * Bb * Hh);  // h0bf,h1bf,ht0bf (zeroed)
  unsigned short* h0bf  = bfmir;
  unsigned short* h1bf  = bfmir + Bb * Hh;
  unsigned short* ht0bf = bfmir + 2 * Bb * Hh;
  unsigned short* ct_bf = allocBF((size_t)Bb * Hh);

  float* enc_h  = allocF((size_t)Ss * Bb * Hh);
  float* states = allocF((size_t)4 * Bb * Hh);   // h0,c0,h1,c1 (zeroed)
  float* h0 = states, *c0 = states + Bb * Hh, *h1 = states + 2 * Bb * Hh,
       * c1 = states + 3 * Bb * Hh;
  float* z    = allocF((size_t)Bb * G4);
  float* tmp  = allocF((size_t)Bb * Hh);
  float* ptb  = allocF(Bb);
  float* pmax = allocF((size_t)Tt * Bb * FG);
  float* psum = allocF((size_t)Tt * Bb * FG);
  float* logZ = allocF((size_t)Tt * Bb);
  float* tl   = allocF((size_t)(Tt - 1) * Bb);

  // --- weight conversion to bf16 (L2-resident working set ~133MB) ---
  auto conv = [&](const float* s, unsigned short* d, size_t n) {
    to_bf16<<<(unsigned)((n + 255) / 256), 256, 0, stream>>>(s, d, (int)n);
  };
  conv(eW_ih0, bWih0e, (size_t)G4 * Ee);
  conv(eW_hh0, bWhh0e, (size_t)G4 * Hh);
  conv(eW_ih1, bWih1e, (size_t)G4 * Hh);
  conv(eW_hh1, bWhh1e, (size_t)G4 * Hh);
  conv(dW_ih0, bWih0d, (size_t)G4 * (Ee + Hh));
  conv(dW_hh0, bWhh0d, (size_t)G4 * Hh);
  conv(dW_ih1, bWih1d, (size_t)G4 * Hh);
  conv(dW_hh1, bWhh1d, (size_t)G4 * Hh);
  conv(W_h2t, bWh2t, (size_t)Hh * Hh);
  conv(W_c2h, bWc2h, (size_t)Hh * 2 * Hh);
  conv(W_fin, bWfin, (size_t)Vv * Hh);

  // --- embeddings (direct to bf16) + state init ---
  gather_emb_bf<<<(Ss * Bb * Ee) / 256, 256, 0, stream>>>(src_tok, emb_src, x_src_bf, Ss * Bb * Ee);
  gather_emb_bf<<<(Tt * Bb * Ee) / 256, 256, 0, stream>>>(tgt_tok, emb_tar, y_tar_bf, Tt * Bb * Ee);
  zero_f32<<<(4 * Bb * Hh) / 256, 256, 0, stream>>>(states, 4 * Bb * Hh);
  zero_f32<<<((3 * Bb * Hh) / 2) / 256, 256, 0, stream>>>((float*)bfmir, (3 * Bb * Hh) / 2);

  auto gemm = [&](const unsigned short* A1, int lda1, const unsigned short* W1, int ldw1, int K1,
                  const unsigned short* A2, int lda2, const unsigned short* W2, int ldw2, int K2,
                  const unsigned short* A3, int lda3, const unsigned short* W3, int ldw3, int K3,
                  const float* bias, float* Cf, unsigned short* Cbf,
                  int ldc, int M, int N, int act) {
    int tiles = (M / 16) * (N / 16);
    gemm_bf16_wmma<<<tiles / 8, 256, 0, stream>>>(
        A1, lda1, W1, ldw1, K1, A2, lda2, W2, ldw2, K2, A3, lda3, W3, ldw3, K3,
        bias, Cf, Cbf, ldc, M / 16, act);
  };

  // --- encoder: 64 steps, 2 layers, packed-sequence masking ---
  for (int t = 0; t < Ss; ++t) {
    const unsigned short* xt = x_src_bf + (size_t)t * Bb * Ee;
    gemm(xt, Ee, bWih0e, Ee, Ee,  h0bf, Hh, bWhh0e, Hh, Hh,
         nullptr, 0, nullptr, 0, 0,  eb0, z, nullptr, G4, Bb, G4, 0);
    lstm_cell<<<(Bb * Hh) / 256, 256, 0, stream>>>(z, h0, c0, h0bf, elen, t, nullptr);
    gemm(h0bf, Hh, bWih1e, Hh, Hh,  h1bf, Hh, bWhh1e, Hh, Hh,
         nullptr, 0, nullptr, 0, 0,  eb1, z, nullptr, G4, Bb, G4, 0);
    lstm_cell<<<(Bb * Hh) / 256, 256, 0, stream>>>(z, h1, c1, h1bf, elen, t,
                                                   enc_h + (size_t)t * Bb * Hh);
  }

  // --- decoder: input-feeding + Gaussian-window local attention ---
  for (int t = 0; t < Tt; ++t) {
    const unsigned short* yt_in = y_tar_bf + (size_t)t * Bb * Ee;
    const unsigned short* htf = (t == 0) ? ht0bf : (dec_out_bf + (size_t)(t - 1) * Bb * Hh);
    // z0 = [y_t, ht] @ dW_ih0^T + h0 @ dW_hh0^T + db0  (3 fused parts)
    gemm(yt_in, Ee, bWih0d, Ee + Hh, Ee,
         htf,   Hh, bWih0d + Ee, Ee + Hh, Hh,
         h0bf,  Hh, bWhh0d, Hh, Hh,
         db0, z, nullptr, G4, Bb, G4, 0);
    lstm_cell<<<(Bb * Hh) / 256, 256, 0, stream>>>(z, h0, c0, h0bf, nullptr, 0, nullptr);
    gemm(h0bf, Hh, bWih1d, Hh, Hh,  h1bf, Hh, bWhh1d, Hh, Hh,
         nullptr, 0, nullptr, 0, 0,  db1, z, nullptr, G4, Bb, G4, 0);
    lstm_cell<<<(Bb * Hh) / 256, 256, 0, stream>>>(z, h1, c1, h1bf, nullptr, 0, nullptr);
    // tmp = tanh(yt @ W_ht2tan^T);  pt = sigmoid(tmp.w)*len
    gemm(h1bf, Hh, bWh2t, Hh, Hh, nullptr, 0, nullptr, 0, 0,
         nullptr, 0, nullptr, 0, 0, nullptr, tmp, nullptr, Hh, Bb, Hh, 1);
    pt_kernel<<<Bb, 256, 0, stream>>>(tmp, w_t2p, elen, ptb);
    attention<<<Bb, 256, 0, stream>>>(h1, enc_h, ptb, elen, ct_bf);
    // ht_new = tanh([ct, yt] @ W_ct2ht^T) -> bf16 dec_out[t] (next input-feed + final GEMM A)
    gemm(ct_bf, Hh, bWc2h, 2 * Hh, Hh,
         h1bf,  Hh, bWc2h + Hh, 2 * Hh, Hh,
         nullptr, 0, nullptr, 0, 0,
         nullptr, nullptr, dec_out_bf + (size_t)t * Bb * Hh, Hh, Bb, Hh, 1);
  }

  // --- final projection + fused streaming log-softmax ---
  final_gemm_partial<<<((Tt * Bb / 16) * FG) / 8, 256, 0, stream>>>(dec_out_bf, bWfin, pmax, psum);
  logz_combine<<<Tt * Bb, 256, 0, stream>>>(pmax, psum, logZ);
  target_logit<<<(Tt - 1) * Bb, 256, 0, stream>>>(dec_out_bf, bWfin, tgt_tok, tl);
  sum_out<<<Bb, 64, 0, stream>>>(tl, logZ, tgt_tok, (float*)d_out);
}
